// MT_RNN_90838558311092
// MI455X (gfx1250) — matrix operand order repaired and checked by reading, not compiled, not run on hardware
//
#include <hip/hip_runtime.h>
#include <hip/hip_bf16.h>
#include <math.h>

// ---------------------------------------------------------------------------
// Problem constants (from reference)
// ---------------------------------------------------------------------------
#define S_LEN 2048
#define BATCH 64
#define XD    64
#define DX    128
#define H_DIM 256
#define DH    128

typedef __attribute__((ext_vector_type(16))) _Float16 v16h;
typedef __attribute__((ext_vector_type(8)))  _Float16 v8h;
typedef __attribute__((ext_vector_type(8)))  float    v8f;

// ---------------------------------------------------------------------------
// Branch-free fast tanh using v_exp_f32 (2^x) + v_rcp_f32.
//   tanh(x) = 1 - 2/(e^(2x)+1);  e^(2x) = 2^(x * 2*log2(e))
// Saturates to +/-1 for large |x| (exp2 -> inf or 0), no divergent branches.
// ---------------------------------------------------------------------------
__device__ __forceinline__ float fast_tanh(float x) {
  float e = __builtin_amdgcn_exp2f(x * 2.885390081777927f);
  return 1.0f - 2.0f * __builtin_amdgcn_rcpf(e + 1.0f);
}
__device__ __forceinline__ float fast_sigmoid(float x) {
  float e = __builtin_amdgcn_exp2f(-x * 1.4426950408889634f);
  return __builtin_amdgcn_rcpf(1.0f + e);
}

// ---------------------------------------------------------------------------
// WMMA fragment helpers (wave32; A/B use this code's contiguous k order, C/D per CDNA5 ISA 7.12.2)
//   A (16x32 f16): lane<16 holds row=lane, K = {kb..kb+15}, kb=0
//                  lane>=16 same row group, kb=16
//   B (32x16 f16): stored pre-swizzled in LDS: frag*512 + lane*16 + i halves
//   C/D (16x16 f32): vgpr r -> M = r + (lane>=16 ? 8 : 0), N = lane&15
// ---------------------------------------------------------------------------
__device__ __forceinline__ v16h mk16(v8h lo, v8h hi) {
  v16h r;
#pragma unroll
  for (int i = 0; i < 8; ++i) { r[i] = lo[i]; r[i + 8] = hi[i]; }
  return r;
}

__device__ __forceinline__ v16h load_a_frag(const _Float16* s, int stride,
                                            int kt, int lane) {
  int row = lane & 15;
  int kb  = kt * 32 + ((lane & 16) ? 16 : 0);
  const _Float16* p = s + row * stride + kb;
  v8h lo = *(const v8h*)(p);
  v8h hi = *(const v8h*)(p + 8);
  return mk16(lo, hi);
}

__device__ __forceinline__ v16h load_b_frag(const _Float16* frags, int fragIdx,
                                            int lane) {
  const _Float16* p = frags + (fragIdx << 9) + (lane << 4);
  v8h lo = *(const v8h*)(p);
  v8h hi = *(const v8h*)(p + 8);
  return mk16(lo, hi);
}

__device__ __forceinline__ v8f wmma_f16(v16h a, v16h b, v8f c) {
  return __builtin_amdgcn_wmma_f32_16x16x32_f16(false, a, false, b,
                                                (short)0, c, false, false);
}

// ---------------------------------------------------------------------------
// Kernel 1: feat[t,b,:] = tanh(x[t,b,:] @ Wx + bx)  -> fp16 workspace
// grid: (S*B)/8 blocks of 256 threads; each block does 8 rows x 128 cols
// ---------------------------------------------------------------------------
__global__ __launch_bounds__(256)
void feat_kernel(const float* __restrict__ x, const float* __restrict__ Wx,
                 const float* __restrict__ bx, _Float16* __restrict__ fws) {
  __shared__ float sx[8][XD];
  const int tid  = threadIdx.x;
  const size_t row0 = (size_t)blockIdx.x * 8;

  for (int i = tid; i < 8 * XD; i += 256) {
    int r = i >> 6, k = i & 63;
    sx[r][k] = x[(row0 + r) * XD + k];
  }
  __syncthreads();

  const int d = tid & 127;
  for (int rr = (tid >> 7); rr < 8; rr += 2) {
    float acc = bx[d];
#pragma unroll 8
    for (int k = 0; k < XD; ++k) acc = fmaf(sx[rr][k], Wx[k * DX + d], acc);
    fws[(row0 + rr) * DX + d] = (_Float16)fast_tanh(acc);
  }
}

// ---------------------------------------------------------------------------
// Kernel 2: persistent RNN scan. 4 blocks x 256 threads (8 waves).
// Block bt owns batch rows [bt*16, bt*16+16). All weights live in LDS (fp16).
// h kept double-buffered in fp16 (WMMA A operand) + fp32 master for the
// leaky blend; only 2 barriers per timestep.
//
// LDS map (halves unless noted):
//   sWhh  frags: KT=8,  NT=16 -> 128*512 = 65536   (B[k][n] = Whh[n][k])
//   sWih  frags: KT=4,  NT=16 ->  64*512 = 32768   (B[k][n] = Wih[n][k])
//   sWhx  frags: KT=8,  NT=8  ->  64*512 = 32768   (B[k][n] = Whx[k][n])
//   sWout frags: KT=4,  NT=4  ->  16*512 =  8192   (B[k][n] = Wout[k][n])
//   sH0/sH1 fp16 [16][264] = 2*4224  (padded stride: conflict-free b128)
//   sF  fp16 [16][136]  = 2176
//   sG  fp16 [16][136]  = 2176
//   sHf32 float [16][260] = 4160 floats
//   sBias[256], sAlpha[256], sBhx[128], sBout[64] floats
// total = 323,584 bytes (< 320 KB WGP LDS)
// ---------------------------------------------------------------------------
#define HS   264   // sH stride (halves)
#define GS   136   // sF/sG stride (halves)
#define HFS  260   // sHf32 stride (floats)

__global__ __launch_bounds__(256)
void rnn_scan_kernel(const _Float16* __restrict__ fws,
                     const float* __restrict__ Wih, const float* __restrict__ bih,
                     const float* __restrict__ Whh, const float* __restrict__ bhh,
                     const float* __restrict__ Whx, const float* __restrict__ bhx,
                     const float* __restrict__ Wout, const float* __restrict__ bout,
                     const float* __restrict__ sigmas,
                     float* __restrict__ out) {
  extern __shared__ char smem[];
  _Float16* sWhh  = (_Float16*)smem;
  _Float16* sWih  = sWhh  + 65536;
  _Float16* sWhx  = sWih  + 32768;
  _Float16* sWout = sWhx  + 32768;
  _Float16* sH0   = sWout + 8192;
  _Float16* sH1   = sH0   + 16 * HS;
  _Float16* sF    = sH1   + 16 * HS;
  _Float16* sG    = sF    + 16 * GS;
  float*    sHf32 = (float*)(sG + 16 * GS);
  float*    sBias = sHf32 + 16 * HFS;
  float*    sAlph = sBias + 256;
  float*    sBhx  = sAlph + 256;
  float*    sBout = sBhx  + 128;

  const int tid  = threadIdx.x;
  const int lane = tid & 31;
  const int wave = tid >> 5;
  const int b0   = blockIdx.x * 16;

  // ---- one-time init: weights -> fp16 swizzled fragments ------------------
  // Whh: out[m][n] = sum_k h[m][k]*Whh[n][k]  => B[k][n] = Whh[n][k]
  for (int o = tid; o < 128 * 512; o += 256) {
    int f = o >> 9, w = o & 511, ln = w >> 4, i = w & 15;
    int kt = f >> 4, nt = f & 15;
    int k = kt * 32 + ((ln < 16) ? i : 16 + i);
    int n = nt * 16 + (ln & 15);
    sWhh[o] = (_Float16)Whh[n * H_DIM + k];
  }
  // Wih^T: B[k][n] = Wih[n][k], k<128, n<256
  for (int o = tid; o < 64 * 512; o += 256) {
    int f = o >> 9, w = o & 511, ln = w >> 4, i = w & 15;
    int kt = f >> 4, nt = f & 15;
    int k = kt * 32 + ((ln < 16) ? i : 16 + i);
    int n = nt * 16 + (ln & 15);
    sWih[o] = (_Float16)Wih[n * DX + k];
  }
  // Whx: B[k][n] = Whx[k][n], k<256, n<128  (KT=8, NT=8)
  for (int o = tid; o < 64 * 512; o += 256) {
    int f = o >> 9, w = o & 511, ln = w >> 4, i = w & 15;
    int kt = f >> 3, nt = f & 7;
    int k = kt * 32 + ((ln < 16) ? i : 16 + i);
    int n = nt * 16 + (ln & 15);
    sWhx[o] = (_Float16)Whx[k * DH + n];
  }
  // Wout: B[k][n] = Wout[k][n], k<128, n<64  (KT=4, NT=4)
  for (int o = tid; o < 16 * 512; o += 256) {
    int f = o >> 9, w = o & 511, ln = w >> 4, i = w & 15;
    int kt = f >> 2, nt = f & 3;
    int k = kt * 32 + ((ln < 16) ? i : 16 + i);
    int n = nt * 16 + (ln & 15);
    sWout[o] = (_Float16)Wout[k * XD + n];
  }
  // h0 = 0 (fp32 master + fp16 buffer 0)
  for (int i = tid; i < 16 * HFS; i += 256) sHf32[i] = 0.0f;
  for (int i = tid; i < 16 * HS;  i += 256) sH0[i] = (_Float16)0.0f;
  // biases / alphas
  if (tid < 256) {
    sBias[tid] = bih[tid] + bhh[tid];
    sAlph[tid] = fast_sigmoid(sigmas[tid & 3]);  // tiled per-unit alpha
  }
  if (tid < 128) sBhx[tid]  = bhx[tid];
  if (tid < 64)  sBout[tid] = bout[tid];
  __syncthreads();

  _Float16* sHcur = sH0;
  _Float16* sHnxt = sH1;

  // ---- sequential scan ----------------------------------------------------
  for (int t = 0; t < S_LEN; ++t) {
    // load feat tile [16][128] for this step (L2-resident fp16)
    {
      int m = tid >> 4, c = tid & 15;
      const _Float16* gp = fws + (((size_t)t * BATCH) + b0 + m) * DX + c * 8;
      *(v8h*)(sF + m * GS + c * 8) = *(const v8h*)gp;
      if (t + 1 < S_LEN) {
        __builtin_prefetch(gp + (size_t)BATCH * DX, 0, 1);  // next step tile
      }
    }

    // Phase B: g = tanh(h @ Whx + bhx)   [16 x 128], wave -> ntile
    {
      const int nt = wave;  // 0..7
      v8f c = {};
#pragma unroll 4
      for (int kt = 0; kt < 8; ++kt) {
        v16h a = load_a_frag(sHcur, HS, kt, lane);
        v16h b = load_b_frag(sWhx, kt * 8 + nt, lane);
        c = wmma_f16(a, b, c);
      }
#pragma unroll
      for (int r = 0; r < 8; ++r) {
        int m = r + ((lane & 16) ? 8 : 0);
        int n = nt * 16 + (lane & 15);
        sG[m * GS + n] = (_Float16)fast_tanh(c[r] + sBhx[n]);
      }
    }
    __syncthreads();  // sG ready; sF ready for Phase D

    // Phase C (waves 0..3): y_t = g @ Wout + bout -> global  [16 x 64]
    if (wave < 4) {
      const int nt = wave;
      v8f c = {};
#pragma unroll
      for (int kt = 0; kt < 4; ++kt) {
        v16h a = load_a_frag(sG, GS, kt, lane);
        v16h b = load_b_frag(sWout, kt * 4 + nt, lane);
        c = wmma_f16(a, b, c);
      }
#pragma unroll
      for (int r = 0; r < 8; ++r) {
        int m = r + ((lane & 16) ? 8 : 0);
        int n = nt * 16 + (lane & 15);
        out[((size_t)t * BATCH + (b0 + m)) * XD + n] = c[r] + sBout[n];
      }
    }

    // Phase D (all waves): pre = f@Wih^T + h@Whh^T + (bih+bhh);
    //   h' = (1-a)*h + a*tanh(pre); write fp32 master + fp16 next buffer
#pragma unroll
    for (int q = 0; q < 2; ++q) {
      const int nt = wave * 2 + q;  // 0..15
      v8f c = {};
#pragma unroll
      for (int kt = 0; kt < 4; ++kt) {
        v16h a = load_a_frag(sF, GS, kt, lane);
        v16h b = load_b_frag(sWih, kt * 16 + nt, lane);
        c = wmma_f16(a, b, c);
      }
#pragma unroll 4
      for (int kt = 0; kt < 8; ++kt) {
        v16h a = load_a_frag(sHcur, HS, kt, lane);
        v16h b = load_b_frag(sWhh, kt * 16 + nt, lane);
        c = wmma_f16(a, b, c);
      }
#pragma unroll
      for (int r = 0; r < 8; ++r) {
        int m = r + ((lane & 16) ? 8 : 0);
        int n = nt * 16 + (lane & 15);
        float hn = fast_tanh(c[r] + sBias[n]);
        float al = sAlph[n];
        float hv = sHf32[m * HFS + n];
        float hx = hv + al * (hn - hv);      // (1-a)*h + a*hn
        sHf32[m * HFS + n] = hx;
        sHnxt[m * HS + n]  = (_Float16)hx;
      }
    }
    __syncthreads();  // sHnxt complete; sF free for next load

    _Float16* tmp = sHcur; sHcur = sHnxt; sHnxt = tmp;
  }
}

// ---------------------------------------------------------------------------
// launch
// ---------------------------------------------------------------------------
extern "C" void kernel_launch(void* const* d_in, const int* in_sizes, int n_in,
                              void* d_out, int out_size, void* d_ws, size_t ws_size,
                              hipStream_t stream) {
  const float* x      = (const float*)d_in[0];
  const float* Wx     = (const float*)d_in[1];
  const float* bx     = (const float*)d_in[2];
  const float* Wih    = (const float*)d_in[3];
  const float* bih    = (const float*)d_in[4];
  const float* Whh    = (const float*)d_in[5];
  const float* bhh    = (const float*)d_in[6];
  const float* Whx    = (const float*)d_in[7];
  const float* bhx    = (const float*)d_in[8];
  const float* Wout   = (const float*)d_in[9];
  const float* bout   = (const float*)d_in[10];
  const float* sigmas = (const float*)d_in[11];
  float* out = (float*)d_out;

  _Float16* fws = (_Float16*)d_ws;  // S*B*DX halves = 32 MB

  // 1) feature extractor over all timesteps
  feat_kernel<<<(S_LEN * BATCH) / 8, 256, 0, stream>>>(x, Wx, bx, fws);

  // 2) persistent WMMA scan: 4 blocks x 16 batch rows, ~316 KB dynamic LDS
  const size_t lds_bytes =
      (size_t)(65536 + 32768 + 32768 + 8192 + 2 * 16 * HS + 2 * 16 * GS) * 2 +
      (size_t)(16 * HFS + 256 + 256 + 128 + 64) * 4;
  rnn_scan_kernel<<<4, 256, lds_bytes, stream>>>(
      fws, Wih, bih, Whh, bhh, Whx, bhx, Wout, bout, sigmas, out);
}
